// VecDGCNN_v2_81999515615312
// MI455X (gfx1250) — matrix operand order repaired and checked by reading, not compiled, not run on hardware
//
#include <hip/hip_runtime.h>
#include <hip/hip_bf16.h>
#include <cstdint>

#define EPSV 1e-6f
#define SLOPE 0.2f
#define SCALE_FACTOR 640.0f

enum : int { BB = 2, NPTS = 2048, HC = 128, KNN_K = 16, NL = 4, NCOLS = 3 * NPTS };

typedef __attribute__((ext_vector_type(16))) __bf16 v16bf;
typedef __attribute__((ext_vector_type(8)))  __bf16 v8bf;
typedef __attribute__((ext_vector_type(8)))  float  v8f;

__device__ __forceinline__ void wait_asynccnt0() {
#if __has_builtin(__builtin_amdgcn_s_wait_asynccnt)
    __builtin_amdgcn_s_wait_asynccnt(0);
#else
    asm volatile("s_wait_asynccnt 0x0" ::: "memory");
#endif
}

// ---------------------------------------------------------------------------
// shared nonlinearity:  out = x + (leaky_relu(<x,kd>) - <x,kd>) * kd,
// kd = k / (||k|| + EPS), per 3-vector
// ---------------------------------------------------------------------------
__device__ __forceinline__ void vec_act3(const float* lin, const float* kv, float* out) {
    float kn = sqrtf(kv[0] * kv[0] + kv[1] * kv[1] + kv[2] * kv[2]) + EPSV;
    float r = 1.0f / kn;
    float kd0 = kv[0] * r, kd1 = kv[1] * r, kd2 = kv[2] * r;
    float dot = lin[0] * kd0 + lin[1] * kd1 + lin[2] * kd2;
    float f = (dot >= 0.0f ? dot : SLOPE * dot) - dot;
    out[0] = lin[0] + f * kd0;
    out[1] = lin[1] + f * kd1;
    out[2] = lin[2] + f * kd2;
}

// ---------------------------------------------------------------------------
// bf16 WMMA GEMM.  C[b][col][ldC] (col-major)  =  A[M][Kd] * Bm[b][Kd][NC]
// Block: 256 threads (8 waves), 128x128 tile, K-step 32, double-buffered LDS.
// A tile staged with global_load_async_to_lds_b128 (ASYNCcnt-tracked DMA);
// B tile staged transposed via b64-packed ds stores.
// Compute: preload all 8 B fragments (one s_wait_dscnt), then 8 back-to-back
// v_wmma_f32_16x16x32_bf16 on independent accumulators.
// ---------------------------------------------------------------------------
#define LDK 40   // padded LDS stride (bf16 elems); 80B rows, 16B-aligned chunks

__global__ __launch_bounds__(256)
void gemm_bf16(const __bf16* __restrict__ A, const __bf16* __restrict__ Bm,
               float* __restrict__ C, int Kd, int NC, int ldC, size_t sB) {
    __shared__ __bf16 As[2][128 * LDK];
    __shared__ __bf16 Bs[2][128 * LDK];
    const int b  = blockIdx.z;
    const int c0 = blockIdx.x * 128;
    const int m0 = blockIdx.y * 128;
    const int t = threadIdx.x;
    const int wave = t >> 5, lane = t & 31;
    const int half = lane >> 4, l16 = lane & 15;
    const __bf16* Ag = A + (size_t)m0 * Kd;
    const __bf16* Bg = Bm + (size_t)b * sB;

    v8f acc[8] = {};

    const int ar = t >> 1, ah = t & 1;   // A stage: row, 16-elem half
    const int kg = t >> 5, cg = t & 31;  // B stage: 4-k group, 4-col group

    // hoisted thread-invariant staging bases
    const uint64_t gaBase = (uint64_t)(uintptr_t)(Ag + (size_t)ar * Kd + ah * 16);
    const uint32_t laBase0 = (uint32_t)(uintptr_t)&As[0][ar * LDK + ah * 16];
    const uint32_t laBase1 = (uint32_t)(uintptr_t)&As[1][ar * LDK + ah * 16];
    const __bf16* BgStage = Bg + (size_t)(kg * 4) * NC + c0 + cg * 4;
    __bf16* BsBase0 = &Bs[0][(cg * 4) * LDK + kg * 4];
    __bf16* BsBase1 = &Bs[1][(cg * 4) * LDK + kg * 4];

    // ---- A tile stage: two per-lane 16B async global->LDS copies ----
    auto stageA = [&](int buf, int k0) {
        uint64_t ga = gaBase + (uint64_t)k0 * 2u;
        uint32_t la = buf ? laBase1 : laBase0;
        asm volatile("global_load_async_to_lds_b128 %0, %1, off\n\t"
                     "global_load_async_to_lds_b128 %2, %3, off"
                     :: "v"(la), "v"(ga), "v"(la + 16u), "v"(ga + 16ull)
                     : "memory");
    };
    // ---- B tile stage: 4k x 4col register transpose, b64 LDS stores ----
    auto stageB = [&](int buf, int k0) {
        const __bf16* p = BgStage + (size_t)k0 * NC;
        __bf16* s = buf ? BsBase1 : BsBase0;
        union { uint2 u; __bf16 h[4]; } row[4];
#pragma unroll
        for (int r = 0; r < 4; r++)
            row[r].u = *reinterpret_cast<const uint2*>(p + (size_t)r * NC);
#pragma unroll
        for (int j = 0; j < 4; j++) {
            union { uint2 u; __bf16 h[4]; } colp;
            colp.h[0] = row[0].h[j]; colp.h[1] = row[1].h[j];
            colp.h[2] = row[2].h[j]; colp.h[3] = row[3].h[j];
            *reinterpret_cast<uint2*>(s + j * LDK) = colp.u;
        }
    };

    stageA(0, 0);
    stageB(0, 0);
    wait_asynccnt0();
    __syncthreads();

    for (int k0 = 0; k0 < Kd; k0 += 32) {
        const int cur = (k0 >> 5) & 1;
        if (k0 + 32 < Kd) { stageA(cur ^ 1, k0 + 32); stageB(cur ^ 1, k0 + 32); }

        // A fragment per CDNA5 16-bit 16x32 layout: K = {8h..8h+7, 16+8h..16+8h+7}
        const __bf16* Ab = &As[cur][(wave * 16 + l16) * LDK];
        v8bf alo = *reinterpret_cast<const v8bf*>(Ab + half * 8);
        v8bf ahi = *reinterpret_cast<const v8bf*>(Ab + 16 + half * 8);
        v16bf afrag = __builtin_shufflevector(alo, ahi, 0, 1, 2, 3, 4, 5, 6, 7, 8, 9, 10, 11, 12, 13, 14, 15);

        // preload all 8 B fragments, then issue the 8 WMMAs back-to-back
        const __bf16* Bb = &Bs[cur][l16 * LDK + half * 8];
        v16bf bfr[8];
#pragma unroll
        for (int ct = 0; ct < 8; ct++) {
            v8bf blo = *reinterpret_cast<const v8bf*>(Bb + (ct * 16) * LDK);
            v8bf bhi = *reinterpret_cast<const v8bf*>(Bb + (ct * 16) * LDK + 16);
            bfr[ct] = __builtin_shufflevector(blo, bhi, 0, 1, 2, 3, 4, 5, 6, 7, 8, 9, 10, 11, 12, 13, 14, 15);
        }
#pragma unroll
        for (int ct = 0; ct < 8; ct++)
            acc[ct] = __builtin_amdgcn_wmma_f32_16x16x32_bf16(false, afrag, false, bfr[ct],
                                                              (short)0, acc[ct], false, false);

        wait_asynccnt0();   // next tile's async A copies have landed in LDS
        __syncthreads();
    }

    // D layout: VGPR j -> M = j + 8*half, N = l16.  Store col-major C[b][col][ldC].
    float* Cb = C + ((size_t)b * NC + c0) * ldC + m0;
#pragma unroll
    for (int ct = 0; ct < 8; ct++) {
        int col = ct * 16 + l16;
#pragma unroll
        for (int j = 0; j < 8; j++) {
            int m = wave * 16 + j + half * 8;
            Cb[(size_t)col * ldC + m] = acc[ct][j];
        }
    }
}

// ---------------------------------------------------------------------------
// weight packing helpers
// ---------------------------------------------------------------------------
__global__ void pack_cast(__bf16* dst, const float* __restrict__ src, int n) {
    int i = blockIdx.x * 256 + threadIdx.x;
    if (i < n) dst[i] = (__bf16)src[i];
}
__global__ void pack_slice(__bf16* dst, const float* __restrict__ src, int ld, int off) {
    int i = blockIdx.x * 256 + threadIdx.x;
    if (i < HC * HC) { int o = i >> 7, c = i & 127; dst[i] = (__bf16)src[o * ld + off + c]; }
}
__global__ void pack_slice_diff(__bf16* dst, const float* __restrict__ src, int ld) {
    int i = blockIdx.x * 256 + threadIdx.x;
    if (i < HC * HC) { int o = i >> 7, c = i & 127; dst[i] = (__bf16)(src[o * ld + HC + c] - src[o * ld + c]); }
}
// col-major f32 product -> row-major bf16 A matrix
__global__ void transpose_bf16(__bf16* dst, const float* __restrict__ srcCM) {
    int i = blockIdx.x * 256 + threadIdx.x;
    if (i < HC * HC) { int o = i >> 7, c = i & 127; dst[i] = (__bf16)srcCM[c * HC + o]; }
}
// D0 = conv0_d @ conv0_w  (128x3)
__global__ void d0_kernel(float* D0, const float* __restrict__ W0, const float* __restrict__ Wd) {
    int o = threadIdx.x;
    for (int c = 0; c < 3; c++) {
        float s = 0.0f;
        for (int j = 0; j < HC; j++) s += Wd[o * HC + j] * W0[j * 3 + c];
        D0[o * 3 + c] = s;
    }
}

// ---------------------------------------------------------------------------
// brute-force KNN, point cloud resident in LDS, per-thread sorted top-16 in LDS
// ---------------------------------------------------------------------------
__global__ __launch_bounds__(256)
void knn_kernel(const float* __restrict__ x, int* __restrict__ idx) {
    __shared__ float ptsS[NPTS * 3];
    __shared__ float bd[256 * KNN_K];
    __shared__ int   bi[256 * KNN_K];
    const int b = blockIdx.y, t = threadIdx.x;
    for (int i = t; i < NPTS; i += 256) {
        ptsS[i * 3 + 0] = x[((size_t)b * 3 + 0) * NPTS + i];
        ptsS[i * 3 + 1] = x[((size_t)b * 3 + 1) * NPTS + i];
        ptsS[i * 3 + 2] = x[((size_t)b * 3 + 2) * NPTS + i];
    }
    for (int k = 0; k < KNN_K; k++) { bd[t * KNN_K + k] = 3.4e38f; bi[t * KNN_K + k] = 0; }
    __syncthreads();
    const int n = blockIdx.x * 256 + t;
    const float q0 = ptsS[n * 3], q1 = ptsS[n * 3 + 1], q2 = ptsS[n * 3 + 2];
    for (int m = 0; m < NPTS; m++) {
        float dx = ptsS[m * 3] - q0, dy = ptsS[m * 3 + 1] - q1, dz = ptsS[m * 3 + 2] - q2;
        float d2 = dx * dx + dy * dy + dz * dz;
        if (d2 < bd[t * KNN_K + KNN_K - 1]) {
            int j = KNN_K - 1;
            while (j > 0 && bd[t * KNN_K + j - 1] > d2) {
                bd[t * KNN_K + j] = bd[t * KNN_K + j - 1];
                bi[t * KNN_K + j] = bi[t * KNN_K + j - 1];
                j--;
            }
            bd[t * KNN_K + j] = d2; bi[t * KNN_K + j] = m;
        }
    }
    for (int k = 0; k < KNN_K; k++) idx[((size_t)b * NPTS + n) * KNN_K + k] = bi[t * KNN_K + k];
}

// ---------------------------------------------------------------------------
// layer 0: cross-feature edge conv + vec_act + mean over K (K-dim of GEMM is 3)
// ---------------------------------------------------------------------------
__global__ void edge0_kernel(const float* __restrict__ x, const int* __restrict__ idx,
                             const float* __restrict__ W0, const float* __restrict__ D0,
                             __bf16* __restrict__ yE, float* __restrict__ gsum) {
    const int n = blockIdx.x, b = blockIdx.y, o = threadIdx.x;
    __shared__ int sidx[KNN_K];
    __shared__ float nbp[KNN_K][3];
    __shared__ float selfp[3];
    if (o < KNN_K) sidx[o] = idx[((size_t)b * NPTS + n) * KNN_K + o];
    if (o < 3) selfp[o] = x[((size_t)b * 3 + o) * NPTS + n];
    __syncthreads();
    if (o < KNN_K) {
        int m = sidx[o];
        for (int v = 0; v < 3; v++) nbp[o][v] = x[((size_t)b * 3 + v) * NPTS + m];
    }
    __syncthreads();
    float xv0 = selfp[0], xv1 = selfp[1], xv2 = selfp[2];
    float xn = sqrtf(xv0 * xv0 + xv1 * xv1 + xv2 * xv2);
    float ir = 1.0f / fmaxf(xn, 1e-12f);
    float xd0 = xv0 * ir, xd1 = xv1 * ir, xd2 = xv2 * ir;
    float a0 = W0[o * 3 + 0], a1 = W0[o * 3 + 1], a2 = W0[o * 3 + 2];
    float d0 = D0[o * 3 + 0], d1 = D0[o * 3 + 1], d2 = D0[o * 3 + 2];
    float acc[3] = {0, 0, 0};
    for (int kk = 0; kk < KNN_K; kk++) {
        float nb0 = nbp[kk][0], nb1 = nbp[kk][1], nb2 = nbp[kk][2];
        float cr[3] = { xd1 * nb2 - xd2 * nb1, xd2 * nb0 - xd0 * nb2, xd0 * nb1 - xd1 * nb0 };
        float df[3] = { nb0 - xv0, nb1 - xv1, nb2 - xv2 };
        float xp[3] = { xv0, xv1, xv2 };
        float lin[3], kv[3], out[3];
        for (int v = 0; v < 3; v++) {
            lin[v] = a0 * cr[v] + a1 * df[v] + a2 * xp[v];
            kv[v]  = d0 * cr[v] + d1 * df[v] + d2 * xp[v];
        }
        vec_act3(lin, kv, out);
        acc[0] += out[0]; acc[1] += out[1]; acc[2] += out[2];
    }
    const float inv = 1.0f / KNN_K;
    __bf16* ye = yE + ((size_t)b * HC + o) * NCOLS;
    for (int v = 0; v < 3; v++) {
        float r = acc[v] * inv;
        ye[v * NPTS + n] = (__bf16)r;
        atomicAdd(&gsum[(b * HC + o) * 3 + v], r);
    }
}

// ---------------------------------------------------------------------------
// layers 1..3 edge: gather factored GEMM outputs, vec_act, mean over K
// T layout: [b][col][512], rows {0:t1, 128:t2, 256:u1, 384:u2}
// ---------------------------------------------------------------------------
__global__ void edge_kernel(const float* __restrict__ T, const int* __restrict__ idx,
                            __bf16* __restrict__ yE, float* __restrict__ gsum) {
    const int n = blockIdx.x, b = blockIdx.y, o = threadIdx.x;
    __shared__ int sidx[KNN_K];
    if (o < KNN_K) sidx[o] = idx[((size_t)b * NPTS + n) * KNN_K + o];
    __syncthreads();
    const float* Tb = T + (size_t)b * NCOLS * 512;
    float t2v[3], u2v[3];
#pragma unroll
    for (int v = 0; v < 3; v++) {
        size_t cn = (size_t)(v * NPTS + n) * 512;
        t2v[v] = Tb[cn + 128 + o];
        u2v[v] = Tb[cn + 384 + o];
    }
    float acc[3] = {0, 0, 0};
    for (int kk = 0; kk < KNN_K; kk++) {
        int m = sidx[kk];
        float lin[3], kv[3], out[3];
#pragma unroll
        for (int v = 0; v < 3; v++) {
            size_t cm = (size_t)(v * NPTS + m) * 512;
            lin[v] = Tb[cm + o]       + t2v[v];
            kv[v]  = Tb[cm + 256 + o] + u2v[v];
        }
        vec_act3(lin, kv, out);
        acc[0] += out[0]; acc[1] += out[1]; acc[2] += out[2];
    }
    const float inv = 1.0f / KNN_K;
    __bf16* ye = yE + ((size_t)b * HC + o) * NCOLS;
    for (int v = 0; v < 3; v++) {
        float r = acc[v] * inv;
        ye[v * NPTS + n] = (__bf16)r;
        atomicAdd(&gsum[(b * HC + o) * 3 + v], r);
    }
}

// ---------------------------------------------------------------------------
// per-layer biases from the global mean: bias1 = Wg2 @ g, bias2 = Wgd @ bias1
// ---------------------------------------------------------------------------
__global__ void bias_kernel(const float* __restrict__ gsum, const float* __restrict__ Wg,
                            const float* __restrict__ Wd, float* __restrict__ bias1,
                            float* __restrict__ bias2) {
    const int b = blockIdx.x, o = threadIdx.x;
    __shared__ float gS[HC][3];
    __shared__ float t1S[HC][3];
    for (int v = 0; v < 3; v++) gS[o][v] = gsum[(b * HC + o) * 3 + v] * (1.0f / NPTS);
    __syncthreads();
    float t1[3] = {0, 0, 0};
    for (int c = 0; c < HC; c++) {
        float w = Wg[o * (2 * HC) + HC + c];
        for (int v = 0; v < 3; v++) t1[v] += w * gS[c][v];
    }
    for (int v = 0; v < 3; v++) { t1S[o][v] = t1[v]; bias1[(b * HC + o) * 3 + v] = t1[v]; }
    __syncthreads();
    float b2[3] = {0, 0, 0};
    for (int j = 0; j < HC; j++) {
        float w = Wd[o * HC + j];
        for (int v = 0; v < 3; v++) b2[v] += w * t1S[j][v];
    }
    for (int v = 0; v < 3; v++) bias2[(b * HC + o) * 3 + v] = b2[v];
}

// ---------------------------------------------------------------------------
// gconv epilogue: G layout [b][col][256], rows {0:lin, 128:k}; writes fbig row
// ---------------------------------------------------------------------------
__global__ void gconv_post(const float* __restrict__ G, const float* __restrict__ bias1,
                           const float* __restrict__ bias2, __bf16* __restrict__ fbig, int layer) {
    const int n = blockIdx.x, b = blockIdx.y, o = threadIdx.x;
    const float* Gb = G + (size_t)b * NCOLS * 256;
    float lin[3], kv[3], out[3];
#pragma unroll
    for (int v = 0; v < 3; v++) {
        size_t cn = (size_t)(v * NPTS + n) * 256;
        lin[v] = Gb[cn + o]       + bias1[(b * HC + o) * 3 + v];
        kv[v]  = Gb[cn + 128 + o] + bias2[(b * HC + o) * 3 + v];
    }
    vec_act3(lin, kv, out);
    __bf16* fp = fbig + ((size_t)b * (4 * HC) + layer * HC + o) * NCOLS;
    for (int v = 0; v < 3; v++) fp[v * NPTS + n] = (__bf16)out[v];
}

// ---------------------------------------------------------------------------
// convc epilogue: 1-channel direction via block reduction, then vec_act and
// atomic accumulation of the N-mean into xgsum
// ---------------------------------------------------------------------------
__global__ void convc_post(const float* __restrict__ Ct, const float* __restrict__ dvec,
                           float* __restrict__ xgsum) {
    const int n = blockIdx.x, b = blockIdx.y, o = threadIdx.x;
    const float* Cb = Ct + (size_t)b * NCOLS * 128;
    float lin[3];
#pragma unroll
    for (int v = 0; v < 3; v++) lin[v] = Cb[(size_t)(v * NPTS + n) * 128 + o];
    __shared__ float rr[HC * 3];
    float w = dvec[o];
    for (int v = 0; v < 3; v++) rr[o * 3 + v] = w * lin[v];
    __syncthreads();
    for (int s = 64; s > 0; s >>= 1) {
        if (o < s) for (int v = 0; v < 3; v++) rr[o * 3 + v] += rr[(o + s) * 3 + v];
        __syncthreads();
    }
    float kraw[3] = { rr[0], rr[1], rr[2] };
    float out[3];
    vec_act3(lin, kraw, out);
    for (int v = 0; v < 3; v++) atomicAdd(&xgsum[(b * HC + o) * 3 + v], out[v]);
}

// ---------------------------------------------------------------------------
// final: cev_normalize, scale, fcinv, v_inv.  Output: [scale(2)][z_so3(768)][v_inv(256)]
// ---------------------------------------------------------------------------
__global__ void final_kernel(const float* __restrict__ xgsum, const float* __restrict__ fcinv,
                             float* __restrict__ out) {
    const int b = blockIdx.x, o = threadIdx.x;
    __shared__ float xgS[HC][3];
    __shared__ float red[HC];
    float xg[3];
    for (int v = 0; v < 3; v++) { xg[v] = xgsum[(b * HC + o) * 3 + v] * (1.0f / NPTS); xgS[o][v] = xg[v]; }
    float no = sqrtf(xg[0] * xg[0] + xg[1] * xg[1] + xg[2] * xg[2]);
    red[o] = no * no; __syncthreads();
    for (int s = 64; s > 0; s >>= 1) { if (o < s) red[o] += red[o + s]; __syncthreads(); }
    float den = sqrtf(red[0]);
    __syncthreads();
    float nn = no / fmaxf(den, 1e-12f);
    float ino = 1.0f / fmaxf(no, 1e-12f);
    float z[3] = { nn * xg[0] * ino, nn * xg[1] * ino, nn * xg[2] * ino };
    red[o] = no; __syncthreads();
    for (int s = 64; s > 0; s >>= 1) { if (o < s) red[o] += red[o + s]; __syncthreads(); }
    float scl = (red[0] * (1.0f / HC)) * SCALE_FACTOR;
    __syncthreads();
    float zd[3] = {0, 0, 0};
    for (int c = 0; c < HC; c++) {
        float w = fcinv[o * HC + c];
        for (int v = 0; v < 3; v++) zd[v] += w * xgS[c][v];
    }
    float m = sqrtf(zd[0] * zd[0] + zd[1] * zd[1] + zd[2] * zd[2]);
    red[o] = m * m; __syncthreads();
    for (int s = 64; s > 0; s >>= 1) { if (o < s) red[o] += red[o + s]; __syncthreads(); }
    float den2 = sqrtf(red[0]);
    float nn2 = m / fmaxf(den2, 1e-12f);
    float im = 1.0f / fmaxf(m, 1e-12f);
    float vi = nn2 * (zd[0] * im * z[0] + zd[1] * im * z[1] + zd[2] * im * z[2]);
    if (o == 0) out[b] = scl;
    for (int v = 0; v < 3; v++) out[2 + (b * HC + o) * 3 + v] = z[v];
    out[2 + 2 * HC * 3 + b * HC + o] = vi;
}

// ---------------------------------------------------------------------------
extern "C" void kernel_launch(void* const* d_in, const int* in_sizes, int n_in,
                              void* d_out, int out_size, void* d_ws, size_t ws_size,
                              hipStream_t stream) {
    (void)in_sizes; (void)n_in; (void)out_size; (void)ws_size;
    const float* x       = (const float*)d_in[0];
    const float* conv0_w = (const float*)d_in[1];
    const float* conv0_d = (const float*)d_in[2];
    const float* conv_w  = (const float*)d_in[3];
    const float* conv_d  = (const float*)d_in[4];
    const float* gconv_w = (const float*)d_in[5];
    const float* gconv_d = (const float*)d_in[6];
    const float* convc_w = (const float*)d_in[7];
    const float* convc_d = (const float*)d_in[8];
    const float* fcinv_w = (const float*)d_in[9];
    float* out = (float*)d_out;

    char* ws = (char*)d_ws;
    size_t off = 0;
    auto alloc = [&](size_t bytes) -> char* {
        char* p = ws + off;
        off = (off + bytes + 255) & ~(size_t)255;
        return p;
    };
    int*    IDX   = (int*)   alloc((size_t)BB * NPTS * KNN_K * 4);
    float*  D0    = (float*) alloc((size_t)HC * 3 * 4);
    float*  GSUM  = (float*) alloc((size_t)BB * HC * 3 * 4);
    float*  BIAS1 = (float*) alloc((size_t)BB * HC * 3 * 4);
    float*  BIAS2 = (float*) alloc((size_t)BB * HC * 3 * 4);
    float*  XGS   = (float*) alloc((size_t)BB * HC * 3 * 4);
    __bf16* YE    = (__bf16*)alloc((size_t)BB * HC * NCOLS * 2);
    float*  TBIG  = (float*) alloc((size_t)BB * NCOLS * 512 * 4);   // also aliased as Gt/Ct
    __bf16* FBIG  = (__bf16*)alloc((size_t)BB * 4 * HC * NCOLS * 2);
    __bf16* ACONV = (__bf16*)alloc((size_t)3 * 4 * HC * HC * 2);    // per layer: [W1; W2-W1; Wd@W1; Wd@(W2-W1)]
    __bf16* AG    = (__bf16*)alloc((size_t)4 * 2 * HC * HC * 2);    // per layer: [Wg1; Wgd@Wg1]
    __bf16* AWC   = (__bf16*)alloc((size_t)HC * 4 * HC * 2);
    __bf16* CDB   = (__bf16*)alloc((size_t)3 * HC * HC * 2);
    __bf16* GDB   = (__bf16*)alloc((size_t)4 * HC * HC * 2);
    float*  PROD  = (float*) alloc((size_t)HC * HC * 4);

    const dim3 b256(256), b128(128);

    // ---- weight packing ----
    pack_cast<<<dim3((HC * 4 * HC + 255) / 256), b256, 0, stream>>>(AWC, convc_w, HC * 4 * HC);
    for (int i = 0; i < 3; i++) {
        __bf16* Ai = ACONV + (size_t)i * 4 * HC * HC;
        pack_slice<<<dim3(64), b256, 0, stream>>>(Ai, conv_w + (size_t)i * HC * 2 * HC, 2 * HC, 0);
        pack_slice_diff<<<dim3(64), b256, 0, stream>>>(Ai + HC * HC, conv_w + (size_t)i * HC * 2 * HC, 2 * HC);
        pack_cast<<<dim3(64), b256, 0, stream>>>(CDB + (size_t)i * HC * HC, conv_d + (size_t)i * HC * HC, HC * HC);
    }
    for (int i = 0; i < 4; i++) {
        __bf16* Ai = AG + (size_t)i * 2 * HC * HC;
        pack_slice<<<dim3(64), b256, 0, stream>>>(Ai, gconv_w + (size_t)i * HC * 2 * HC, 2 * HC, 0);
        pack_cast<<<dim3(64), b256, 0, stream>>>(GDB + (size_t)i * HC * HC, gconv_d + (size_t)i * HC * HC, HC * HC);
    }
    // ---- on-device weight-product GEMMs (128x128x128, WMMA) ----
    for (int i = 0; i < 3; i++) {
        __bf16* Ai = ACONV + (size_t)i * 4 * HC * HC;
        gemm_bf16<<<dim3(1, 1, 1), b256, 0, stream>>>(CDB + (size_t)i * HC * HC, Ai, PROD, HC, HC, HC, 0);
        transpose_bf16<<<dim3(64), b256, 0, stream>>>(Ai + 2 * HC * HC, PROD);
        gemm_bf16<<<dim3(1, 1, 1), b256, 0, stream>>>(CDB + (size_t)i * HC * HC, Ai + HC * HC, PROD, HC, HC, HC, 0);
        transpose_bf16<<<dim3(64), b256, 0, stream>>>(Ai + 3 * HC * HC, PROD);
    }
    for (int i = 0; i < 4; i++) {
        __bf16* Ai = AG + (size_t)i * 2 * HC * HC;
        gemm_bf16<<<dim3(1, 1, 1), b256, 0, stream>>>(GDB + (size_t)i * HC * HC, Ai, PROD, HC, HC, HC, 0);
        transpose_bf16<<<dim3(64), b256, 0, stream>>>(Ai + HC * HC, PROD);
    }
    d0_kernel<<<dim3(1), b128, 0, stream>>>(D0, conv0_w, conv0_d);

    // ---- KNN graph ----
    knn_kernel<<<dim3(NPTS / 256, BB), b256, 0, stream>>>(x, IDX);

    // ---- layer 0 ----
    hipMemsetAsync(GSUM, 0, (size_t)BB * HC * 3 * 4, stream);
    edge0_kernel<<<dim3(NPTS, BB), b128, 0, stream>>>(x, IDX, conv0_w, D0, YE, GSUM);
    bias_kernel<<<dim3(BB), b128, 0, stream>>>(GSUM, gconv_w, gconv_d, BIAS1, BIAS2);
    gemm_bf16<<<dim3(NCOLS / 128, 2, BB), b256, 0, stream>>>(AG, YE, TBIG, HC, NCOLS, 256, (size_t)HC * NCOLS);
    gconv_post<<<dim3(NPTS, BB), b128, 0, stream>>>(TBIG, BIAS1, BIAS2, FBIG, 0);

    // ---- layers 1..3 ----
    for (int i = 1; i < 4; i++) {
        __bf16* Ai = ACONV + (size_t)(i - 1) * 4 * HC * HC;
        gemm_bf16<<<dim3(NCOLS / 128, 4, BB), b256, 0, stream>>>(
            Ai, FBIG + (size_t)(i - 1) * HC * NCOLS, TBIG, HC, NCOLS, 512, (size_t)4 * HC * NCOLS);
        hipMemsetAsync(GSUM, 0, (size_t)BB * HC * 3 * 4, stream);
        edge_kernel<<<dim3(NPTS, BB), b128, 0, stream>>>(TBIG, IDX, YE, GSUM);
        bias_kernel<<<dim3(BB), b128, 0, stream>>>(GSUM, gconv_w + (size_t)i * HC * 2 * HC,
                                                   gconv_d + (size_t)i * HC * HC, BIAS1, BIAS2);
        gemm_bf16<<<dim3(NCOLS / 128, 2, BB), b256, 0, stream>>>(
            AG + (size_t)i * 2 * HC * HC, YE, TBIG, HC, NCOLS, 256, (size_t)HC * NCOLS);
        gconv_post<<<dim3(NPTS, BB), b128, 0, stream>>>(TBIG, BIAS1, BIAS2, FBIG, i);
    }

    // ---- head ----
    hipMemsetAsync(XGS, 0, (size_t)BB * HC * 3 * 4, stream);
    gemm_bf16<<<dim3(NCOLS / 128, 1, BB), b256, 0, stream>>>(AWC, FBIG, TBIG, 4 * HC, NCOLS, 128,
                                                             (size_t)4 * HC * NCOLS);
    convc_post<<<dim3(NPTS, BB), b128, 0, stream>>>(TBIG, convc_d, XGS);
    final_kernel<<<dim3(BB), b128, 0, stream>>>(XGS, fcinv_w, out);
}
